// EncoderDecoder_41266045780930
// MI455X (gfx1250) — matrix operand order repaired and checked by reading, not compiled, not run on hardware
//
#include <hip/hip_runtime.h>
#include <hip/hip_bf16.h>
#include <math.h>

#define HID 256
#define NGATE 1024       // 4*HID
#define ENC_STEPS 19
#define DEC_STEPS 29

typedef __attribute__((ext_vector_type(16))) _Float16 v16h;
typedef __attribute__((ext_vector_type(8)))  _Float16 v8h;
typedef __attribute__((ext_vector_type(8)))  float    v8f;

union AFrag { v16h v; v8h h[2]; };

__device__ __forceinline__ float sigmoidf_(float x) { return 1.0f / (1.0f + expf(-x)); }

// ---------------------------------------------------------------------------
// Pack weights [N=1024, K=256] f32 (row-major N-major) into WMMA B-fragment
// layout, f16:  Bpack[((nt*8 + kc)*32 + lane)*16 + j]
//   lane<16 : column n = nt*16+lane,    K = kc*32 + j        (j = 0..15)
//   lane>=16: column n = nt*16+lane-16, K = kc*32 + 16 + j
// Encoder packs Whh_e; decoder packs (Wih_d + Whh_d) (x==h algebraic fold).
// ---------------------------------------------------------------------------
__global__ void pack_weights_kernel(const float* __restrict__ Wenc,
                                    const float* __restrict__ Wd1,
                                    const float* __restrict__ Wd2,
                                    _Float16* __restrict__ BpackE,
                                    _Float16* __restrict__ BpackD) {
    int idx  = blockIdx.x * blockDim.x + threadIdx.x;   // 0 .. 262143
    int j    = idx & 15;
    int lane = (idx >> 4) & 31;
    int kc   = (idx >> 9) & 7;
    int nt   = idx >> 12;                               // 0 .. 63
    int n    = nt * 16 + (lane & 15);
    int k    = kc * 32 + ((lane >> 4) ? 16 : 0) + j;
    int src  = n * HID + k;
    BpackE[idx] = (_Float16)Wenc[src];
    BpackD[idx] = (_Float16)(Wd1[src] + Wd2[src]);
}

__global__ void prep_bias_kernel(const float* __restrict__ bih_e,
                                 const float* __restrict__ bhh_e,
                                 const float* __restrict__ bih_d,
                                 const float* __restrict__ bhh_d,
                                 float* __restrict__ biasE,
                                 float* __restrict__ biasD) {
    int i = blockIdx.x * blockDim.x + threadIdx.x;      // 0..1023
    if (i < NGATE) {
        biasE[i] = bih_e[i] + bhh_e[i];
        biasD[i] = bih_d[i] + bhh_d[i];
    }
}

__global__ void zero_state_kernel(_Float16* __restrict__ h16,
                                  float* __restrict__ c, size_t n) {
    size_t i = (size_t)blockIdx.x * blockDim.x + threadIdx.x;
    if (i < n) { h16[i] = (_Float16)0.0f; c[i] = 0.0f; }
}

__global__ void zero_c_kernel(float* __restrict__ c, size_t n) {
    size_t i = (size_t)blockIdx.x * blockDim.x + threadIdx.x;
    if (i < n) c[i] = 0.0f;
}

// ---------------------------------------------------------------------------
// Fused LSTM step:
//   gates[16, 1024] = h16[m0:m0+16, :256] @ Bpack (+ bias, + x@WihX.T if enc)
//   then i,f,g,o nonlinearities, c/h update in place, optional fused
//   out = relu(h) @ Wlin.T + blin projection for this step's output slot.
// Block: 256 threads = 8 waves. Grid: B/16. Each block owns 16 batch rows
// exclusively (reads+writes only its own h/c rows -> race-free in place).
// Wave w owns N-tiles nt = g*16 + w*2 + u (g=0..3 gate, u=0..1), so each
// lane holds all 4 gates of hidden units n = w*32 + u*16 + ln in registers.
// ---------------------------------------------------------------------------
template <bool IS_ENC, bool PROJECT>
__global__ __launch_bounds__(256)
void lstm_step_kernel(const _Float16* __restrict__ Bpack,  // packed weights
                      const float*    __restrict__ bias,   // [1024] combined
                      const float*    __restrict__ WihX,   // [1024,2] (enc only)
                      const float*    __restrict__ x,      // [B,2]    (enc only)
                      _Float16*       __restrict__ h16,    // [B,256] state (f16)
                      float*          __restrict__ c,      // [B,256] state (f32)
                      const float*    __restrict__ Wlin,   // [2,256]
                      const float*    __restrict__ blin,   // [2]
                      float*          __restrict__ outSlot)// [B,2] for this step
{
    __shared__ float spart[16][128][2];   // projection partials

    const int wave = threadIdx.x >> 5;
    const int lane = threadIdx.x & 31;
    const int lh   = lane >> 4;           // 0 / 1 (K-half for A, M-half for C)
    const int ln   = lane & 15;
    const int m0   = blockIdx.x * 16;

    // ---------------- GEMM: acc[t] += A(16x32) x B(32x16) over K=256 -------
    v8f acc[8];
#pragma unroll
    for (int t = 0; t < 8; ++t) { v8f z = {0,0,0,0,0,0,0,0}; acc[t] = z; }

    const _Float16* __restrict__ aRow = h16 + (size_t)(m0 + ln) * HID;

#pragma unroll
    for (int kc = 0; kc < 8; ++kc) {
        const int ko = kc * 32;
        AFrag a;
        // A-frag k order used here (contiguous, same as Bpack): lane<16 rows M=ln K=ko+0..15
        //                               lane>=16 rows M=ln K=ko+16..31
        a.h[0] = *(const v8h*)(aRow + ko + lh * 16);
        a.h[1] = *(const v8h*)(aRow + ko + lh * 16 + 8);
#pragma unroll
        for (int t = 0; t < 8; ++t) {
            const int nt = (t >> 1) * 16 + wave * 2 + (t & 1);
            const v16h bf = *(const v16h*)(Bpack +
                ((size_t)((nt * 8 + kc) * 32 + lane) << 4));
            acc[t] = __builtin_amdgcn_wmma_f32_16x16x32_f16(
                false, a.v, false, bf, (short)0, acc[t], false, false);
        }
    }

    // ---------------- Epilogue: gates -> c,h update (all in registers) -----
    // lane holds hidden units nA = wave*32+ln (u=0), nB = wave*32+16+ln (u=1),
    // for rows M = lh*8 + r (r=0..7).  acc[g*2+u][r] is gate g pre-act.
    float bias_l[2][4];
    float wx_l[2][4][2];
    float wl0[2], wl1[2];
#pragma unroll
    for (int u = 0; u < 2; ++u) {
        const int n = wave * 32 + u * 16 + ln;
#pragma unroll
        for (int g = 0; g < 4; ++g) {
            const int N = g * HID + n;
            bias_l[u][g] = bias[N];
            if (IS_ENC) {
                wx_l[u][g][0] = WihX[N * 2 + 0];
                wx_l[u][g][1] = WihX[N * 2 + 1];
            }
        }
        if (PROJECT) { wl0[u] = Wlin[n]; wl1[u] = Wlin[HID + n]; }
    }

#pragma unroll
    for (int r = 0; r < 8; ++r) {
        const int M = lh * 8 + r;
        const size_t row = (size_t)(m0 + M);
        float x0 = 0.0f, x1 = 0.0f;
        if (IS_ENC) { x0 = x[row * 2 + 0]; x1 = x[row * 2 + 1]; }
        float p0 = 0.0f, p1 = 0.0f;
#pragma unroll
        for (int u = 0; u < 2; ++u) {
            const int n = wave * 32 + u * 16 + ln;
            float pre[4];
#pragma unroll
            for (int g = 0; g < 4; ++g) {
                pre[g] = acc[g * 2 + u][r] + bias_l[u][g];
                if (IS_ENC) pre[g] += x0 * wx_l[u][g][0] + x1 * wx_l[u][g][1];
            }
            const float iv = sigmoidf_(pre[0]);
            const float fv = sigmoidf_(pre[1]);
            const float gv = tanhf(pre[2]);
            const float ov = sigmoidf_(pre[3]);
            const size_t idx = row * HID + n;
            const float cn = fv * c[idx] + iv * gv;
            const float hn = ov * tanhf(cn);
            c[idx]   = cn;
            h16[idx] = (_Float16)hn;
            if (PROJECT) {
                const float hr = fmaxf(hn, 0.0f);
                p0 += hr * wl0[u];
                p1 += hr * wl1[u];
            }
        }
        if (PROJECT) {
            spart[M][wave * 16 + ln][0] = p0;
            spart[M][wave * 16 + ln][1] = p1;
        }
    }

    if (PROJECT) {
        __syncthreads();
        const int tid = threadIdx.x;
        if (tid < 32) {               // deterministic tree-free serial reduce
            const int rr = tid >> 1, oo = tid & 1;
            float s = blin[oo];
            for (int q = 0; q < 128; ++q) s += spart[rr][q][oo];
            outSlot[((size_t)(m0 + rr)) * 2 + oo] = s;
        }
    }
}

// ---------------------------------------------------------------------------
extern "C" void kernel_launch(void* const* d_in, const int* in_sizes, int n_in,
                              void* d_out, int out_size, void* d_ws, size_t ws_size,
                              hipStream_t stream) {
    const float* inp   = (const float*)d_in[0];   // [19, B, 2]
    const float* Wih_e = (const float*)d_in[1];   // [1024, 2]
    const float* Whh_e = (const float*)d_in[2];   // [1024, 256]
    const float* bih_e = (const float*)d_in[3];
    const float* bhh_e = (const float*)d_in[4];
    const float* Wih_d = (const float*)d_in[5];   // [1024, 256]
    const float* Whh_d = (const float*)d_in[6];   // [1024, 256]
    const float* bih_d = (const float*)d_in[7];
    const float* bhh_d = (const float*)d_in[8];
    const float* Wlin  = (const float*)d_in[9];   // [2, 256]
    const float* blin  = (const float*)d_in[10];  // [2]
    float* out = (float*)d_out;                   // [30, B, 2]

    const size_t B = (size_t)in_sizes[0] / (ENC_STEPS * 2);   // 16384

    // workspace carving (all offsets 512B-aligned)
    uint8_t* ws = (uint8_t*)d_ws;
    _Float16* BpackE = (_Float16*)(ws + 0);
    _Float16* BpackD = (_Float16*)(ws + (512u << 10));
    float*    biasE  = (float*)(ws + (1024u << 10));
    float*    biasD  = (float*)(ws + (1024u << 10) + 4096);
    _Float16* h16    = (_Float16*)(ws + (2048u << 10));
    float*    cbuf   = (float*)(ws + (2048u << 10) + B * HID * sizeof(_Float16));

    const size_t nState = B * HID;

    // prep
    pack_weights_kernel<<<(64 * 8 * 32 * 16) / 256, 256, 0, stream>>>(
        Whh_e, Wih_d, Whh_d, BpackE, BpackD);
    prep_bias_kernel<<<4, 256, 0, stream>>>(bih_e, bhh_e, bih_d, bhh_d, biasE, biasD);
    zero_state_kernel<<<(unsigned)((nState + 255) / 256), 256, 0, stream>>>(h16, cbuf, nState);

    const unsigned grid = (unsigned)(B / 16);

    // encoder: 19 steps, project h_enc into out[0] on the last step
    for (int t = 0; t < ENC_STEPS; ++t) {
        const float* xt = inp + (size_t)t * B * 2;
        if (t == ENC_STEPS - 1) {
            lstm_step_kernel<true, true><<<grid, 256, 0, stream>>>(
                BpackE, biasE, Wih_e, xt, h16, cbuf, Wlin, blin, out);
        } else {
            lstm_step_kernel<true, false><<<grid, 256, 0, stream>>>(
                BpackE, biasE, Wih_e, xt, h16, cbuf, Wlin, blin, nullptr);
        }
    }

    // decoder restarts cell state at zero (reference scans with (h_enc, c0))
    zero_c_kernel<<<(unsigned)((nState + 255) / 256), 256, 0, stream>>>(cbuf, nState);

    // decoder: 29 steps, every step projects into out[k+1]
    for (int k = 0; k < DEC_STEPS; ++k) {
        lstm_step_kernel<false, true><<<grid, 256, 0, stream>>>(
            BpackD, biasD, nullptr, nullptr, h16, cbuf, Wlin, blin,
            out + (size_t)(k + 1) * B * 2);
    }
}
